// HybridLSTM_52647709114578
// MI455X (gfx1250) — compile-verified, matrix operand order checked by reading, not run on hardware
//
#include <hip/hip_runtime.h>
#include <hip/hip_bf16.h>

typedef __attribute__((ext_vector_type(16))) _Float16 v16h;
typedef __attribute__((ext_vector_type(8)))  _Float16 v8h;
typedef __attribute__((ext_vector_type(8)))  float    v8f;

union HV { v16h v16; v8h v8[2]; };

#define T_STEPS 256
#define HID 50
#define KPAD 64      // 50 h + x + 1 + zero pad
#define NPAD 256     // 4 gates x 64

#define LOG2E 1.4426950408889634f

// branchless fast sigmoid: rcp(1 + exp2(-x*log2e)) — hw exp2/rcp saturate correctly
__device__ __forceinline__ float sigm(float x) {
  return __builtin_amdgcn_rcpf(1.0f + __builtin_amdgcn_exp2f(-LOG2E * x));
}

#if __has_builtin(__builtin_amdgcn_tanhf)
__device__ __forceinline__ float fast_tanh(float x) { return __builtin_amdgcn_tanhf(x); }
#else
// tanh(x) = 2*sigmoid(2x) - 1, branchless
__device__ __forceinline__ float fast_tanh(float x) {
  return __builtin_amdgcn_rcpf(0.5f + 0.5f * __builtin_amdgcn_exp2f(-2.0f * LOG2E * x)) - 1.0f;
}
#endif

// apply generic complex 2x2 gate on `wire` (wire 0 = MSB of 4-bit index)
__device__ __forceinline__ void apply1q(float* re, float* im, int wire,
                                        float u00r, float u00i, float u01r, float u01i,
                                        float u10r, float u10i, float u11r, float u11i) {
  const int mask = 1 << (3 - wire);
#pragma unroll
  for (int idx = 0; idx < 16; ++idx) {
    if (idx & mask) continue;
    const int jj = idx | mask;
    float ar = re[idx], ai = im[idx], br = re[jj], bi = im[jj];
    re[idx] = u00r * ar - u00i * ai + u01r * br - u01i * bi;
    im[idx] = u00r * ai + u00i * ar + u01r * bi + u01i * br;
    re[jj]  = u10r * ar - u10i * ai + u11r * br - u11i * bi;
    im[jj]  = u10r * ai + u10i * ar + u11r * bi + u11i * br;
  }
}

__global__ __launch_bounds__(128) void hybrid_lstm_qnn_kernel(
    const float* __restrict__ x,     // [B, 256, 1]
    const float* __restrict__ W_ih,  // [200, 1]
    const float* __restrict__ W_hh,  // [200, 50]
    const float* __restrict__ b_ih,  // [200]
    const float* __restrict__ b_hh,  // [200]
    const float* __restrict__ Wp,    // [4, 50]
    const float* __restrict__ bp,    // [4]
    const float* __restrict__ qw,    // [2, 4, 3]
    const float* __restrict__ Wo,    // [1, 4]
    const float* __restrict__ bo,    // [1]
    float* __restrict__ out)         // [B, 1]
{
  __shared__ __align__(16) _Float16 Blds[NPAD][KPAD];     // 32 KB: [W_hh | W_ih | b | 0], n-major
  __shared__ __align__(16) _Float16 Hlds[2][16][KPAD];    // 4 KB double-buffered A operand
  __shared__ __align__(16) float    Hf32[16][KPAD];       // 4 KB final h for the head

  const int tid = threadIdx.x;
  const int rowbase = blockIdx.x * 16;

  // ---- stage B image: B[k][n] with n-major storage Blds[n][k] ----
  for (int idx = tid; idx < NPAD * KPAD; idx += 128) {
    const int n = idx >> 6, k = idx & 63;
    const int gi = n >> 6, j = n & 63;
    float val = 0.f;
    if (j < HID) {
      const int g = gi * HID + j;          // torch gate row (i,f,g,o order)
      if (k < HID)       val = W_hh[g * HID + k];
      else if (k == 50)  val = W_ih[g];               // IN == 1
      else if (k == 51)  val = b_ih[g] + b_hh[g];
    }
    Blds[n][k] = (_Float16)val;
  }
  // ---- init both h buffers: h=0, slot50=x_0 (buf0), slot51=1, pad=0 ----
  for (int idx = tid; idx < 2 * 16 * KPAD; idx += 128) {
    const int b = idx >> 10, m = (idx >> 6) & 15, k = idx & 63;
    Hlds[b][m][k] = (k == 51) ? (_Float16)1.0f : (_Float16)0.0f;
  }
  if (tid < 16) Hlds[0][tid][50] = (_Float16)x[(size_t)(rowbase + tid) * T_STEPS];
  __syncthreads();

  const int w    = tid >> 5;         // wave id 0..3
  const int l    = tid & 31;         // lane
  const int ncol = l & 15;           // N within 16x16 tile / column of C
  const int krel = (l & 16) ? 8 : 0; // K sub-block select for A/B fragments
  const int row  = l & 15;           // A-matrix M row
  const int mbs  = (l & 16) ? 8 : 0; // C-matrix row base for this lane
  const int j    = 16 * w + ncol;    // hidden column owned by this lane

  // ---- hoist B fragments (loop invariant): bf[gate][kblock] ----
  v16h bf[4][2];
#pragma unroll
  for (int q = 0; q < 4; ++q) {
    const int n = 64 * q + 16 * w + ncol;
#pragma unroll
    for (int kb = 0; kb < 2; ++kb) {
      HV t;
      t.v8[0] = *(const v8h*)&Blds[n][kb * 32 + krel];
      t.v8[1] = *(const v8h*)&Blds[n][kb * 32 + 16 + krel];
      bf[q][kb] = t.v16;
    }
  }

  float cst[8], hout[8];
#pragma unroll
  for (int v = 0; v < 8; ++v) { cst[v] = 0.f; hout[v] = 0.f; }

  // ================= recurrent loop =================
  for (int t = 0; t < T_STEPS; ++t) {
    const int rb = t & 1, wb = rb ^ 1;
    // A fragments (16x32 f16 layout), two K-blocks
    HV ta, tb;
    ta.v8[0] = *(const v8h*)&Hlds[rb][row][krel];
    ta.v8[1] = *(const v8h*)&Hlds[rb][row][16 + krel];
    tb.v8[0] = *(const v8h*)&Hlds[rb][row][32 + krel];
    tb.v8[1] = *(const v8h*)&Hlds[rb][row][48 + krel];
    const v16h a0 = ta.v16, a1 = tb.v16;

    v8f zi = {}, zf = {}, zg = {}, zo = {};
    zi = __builtin_amdgcn_wmma_f32_16x16x32_f16(false, a0, false, bf[0][0], (short)0, zi, false, false);
    zi = __builtin_amdgcn_wmma_f32_16x16x32_f16(false, a1, false, bf[0][1], (short)0, zi, false, false);
    zf = __builtin_amdgcn_wmma_f32_16x16x32_f16(false, a0, false, bf[1][0], (short)0, zf, false, false);
    zf = __builtin_amdgcn_wmma_f32_16x16x32_f16(false, a1, false, bf[1][1], (short)0, zf, false, false);
    zg = __builtin_amdgcn_wmma_f32_16x16x32_f16(false, a0, false, bf[2][0], (short)0, zg, false, false);
    zg = __builtin_amdgcn_wmma_f32_16x16x32_f16(false, a1, false, bf[2][1], (short)0, zg, false, false);
    zo = __builtin_amdgcn_wmma_f32_16x16x32_f16(false, a0, false, bf[3][0], (short)0, zo, false, false);
    zo = __builtin_amdgcn_wmma_f32_16x16x32_f16(false, a1, false, bf[3][1], (short)0, zo, false, false);

#pragma unroll
    for (int v = 0; v < 8; ++v) {
      const float ig = sigm(zi[v]);
      const float fg = sigm(zf[v]);
      const float gg = fast_tanh(zg[v]);
      const float og = sigm(zo[v]);
      cst[v]  = fg * cst[v] + ig * gg;
      hout[v] = og * fast_tanh(cst[v]);
    }
    if (j < HID) {
#pragma unroll
      for (int v = 0; v < 8; ++v)
        Hlds[wb][mbs + v][j] = (_Float16)hout[v];
    }
    if (tid < 16 && (t + 1) < T_STEPS)
      Hlds[wb][tid][50] = (_Float16)x[(size_t)(rowbase + tid) * T_STEPS + (t + 1)];
    __syncthreads();
  }

  // ---- publish final h (f32) ----
#pragma unroll
  for (int v = 0; v < 8; ++v) Hf32[mbs + v][j] = hout[v];
  __syncthreads();

  // ================= per-sample head (lanes 0..15 of the block) =================
  if (tid < 16) {
    float hs[HID];
#pragma unroll
    for (int k = 0; k < HID; ++k) hs[k] = Hf32[tid][k];

    float ang[4];
#pragma unroll
    for (int q = 0; q < 4; ++q) {
      float a = bp[q];
      for (int k = 0; k < HID; ++k) a += Wp[q * HID + k] * hs[k];
      ang[q] = fast_tanh(a) * 1.5707963267948966f;
    }

    float re[16], im[16];
#pragma unroll
    for (int k = 0; k < 16; ++k) { re[k] = 0.f; im[k] = 0.f; }
    re[0] = 1.f;

    // AngleEmbedding: RX(ang[w]) per wire
#pragma unroll
    for (int wi = 0; wi < 4; ++wi) {
      const float c = __cosf(0.5f * ang[wi]);
      const float s = __sinf(0.5f * ang[wi]);
      apply1q(re, im, wi, c, 0.f, 0.f, -s, 0.f, -s, c, 0.f);
    }
    // StronglyEntanglingLayers
#pragma unroll
    for (int ll = 0; ll < 2; ++ll) {
#pragma unroll
      for (int wi = 0; wi < 4; ++wi) {
        const float phi = qw[(ll * 4 + wi) * 3 + 0];
        const float th  = qw[(ll * 4 + wi) * 3 + 1];
        const float om  = qw[(ll * 4 + wi) * 3 + 2];
        const float ct = __cosf(0.5f * th), st = __sinf(0.5f * th);
        const float ap = 0.5f * (phi + om), am = 0.5f * (phi - om);
        apply1q(re, im, wi,
                ct * __cosf(ap), -ct * __sinf(ap),
               -st * __cosf(am), -st * __sinf(am),
                st * __cosf(am), -st * __sinf(am),
                ct * __cosf(ap),  ct * __sinf(ap));
      }
      const int r = ll + 1;  // ranges [1, 2]
#pragma unroll
      for (int wi = 0; wi < 4; ++wi) {
        const int cm = 1 << (3 - wi);
        const int tm = 1 << (3 - ((wi + r) & 3));
#pragma unroll
        for (int idx = 0; idx < 16; ++idx) {
          if ((idx & cm) && !(idx & tm)) {
            const int p = idx | tm;
            float tr = re[idx]; re[idx] = re[p]; re[p] = tr;
            float ti = im[idx]; im[idx] = im[p]; im[p] = ti;
          }
        }
      }
    }
    // <Z_w> readout + linear output
    float res = bo[0];
#pragma unroll
    for (int wi = 0; wi < 4; ++wi) {
      const int m = 1 << (3 - wi);
      float ev = 0.f;
#pragma unroll
      for (int idx = 0; idx < 16; ++idx) {
        const float p = re[idx] * re[idx] + im[idx] * im[idx];
        ev += (idx & m) ? -p : p;
      }
      res += ev * Wo[wi];
    }
    out[rowbase + tid] = res;
  }
}

extern "C" void kernel_launch(void* const* d_in, const int* in_sizes, int n_in,
                              void* d_out, int out_size, void* d_ws, size_t ws_size,
                              hipStream_t stream) {
  const float* x    = (const float*)d_in[0];
  const float* W_ih = (const float*)d_in[1];
  const float* W_hh = (const float*)d_in[2];
  const float* b_ih = (const float*)d_in[3];
  const float* b_hh = (const float*)d_in[4];
  const float* Wp   = (const float*)d_in[5];
  const float* bp   = (const float*)d_in[6];
  const float* qw   = (const float*)d_in[7];
  const float* Wo   = (const float*)d_in[8];
  const float* bo   = (const float*)d_in[9];
  float* out = (float*)d_out;

  const int B = in_sizes[0] / T_STEPS;  // 4096
  dim3 grid(B / 16), block(128);
  hybrid_lstm_qnn_kernel<<<grid, block, 0, stream>>>(
      x, W_ih, W_hh, b_ih, b_hh, Wp, bp, qw, Wo, bo, out);
}
